// MultiHeadAttentionLayer_5832565588424
// MI455X (gfx1250) — matrix operand order repaired and checked by reading, not compiled, not run on hardware
//
#include <hip/hip_runtime.h>
#include <hip/hip_bf16.h>
#include <cstdint>
#include <cstddef>

#define DINL __device__ __forceinline__

typedef __attribute__((ext_vector_type(16))) _Float16 v16h;
typedef __attribute__((ext_vector_type(8)))  _Float16 v8h;
typedef __attribute__((ext_vector_type(8)))  float    v8f;
typedef __attribute__((ext_vector_type(4)))  unsigned int v4u;
typedef __attribute__((ext_vector_type(4)))  int      v4i;
typedef __attribute__((ext_vector_type(8)))  int      v8i;

// Problem constants
constexpr int kS   = 2048;
constexpr int kHID = 1024;
constexpr int kHD  = 64;
constexpr int kH   = 16;
constexpr int kB   = 4;
constexpr int kM   = kB * kS;                  // 8192 flattened rows
constexpr size_t kQKV = (size_t)kM * kHID;     // 8,388,608 elements

// ---------------------------------------------------------------------------
// Tensor Data Mover support (CDNA5): descriptor per CDNA5 ISA.
// ---------------------------------------------------------------------------
#if defined(__has_builtin)
#if __has_builtin(__builtin_amdgcn_tensor_load_to_lds) && \
    __has_builtin(__builtin_amdgcn_s_wait_tensorcnt)
#define USE_TDM 1
#endif
#endif
#ifndef USE_TDM
#define USE_TDM 0
#endif

#if __has_include(<hip/amd_detail/amd_gfx1250_TDM.h>)
#define TDM_6ARG 1
#else
#define TDM_6ARG 0
#endif

#if USE_TDM
// 2D tile load, 2-byte elements: tile_d1 rows of tile_d0 elements, row stride
// stride0 (elements), compacted into LDS at lds_byte_off.
DINL void tdm_load_tile_f16(unsigned lds_byte_off, const void* gptr,
                            unsigned tensor_d0, unsigned tensor_d1,
                            unsigned tile_d0, unsigned tile_d1,
                            unsigned stride0) {
  const unsigned long long ga = (unsigned long long)(uintptr_t)gptr;
  v4u g0;
  g0[0] = 1u;                                        // count=1, user mode
  g0[1] = lds_byte_off;                              // lds_addr
  g0[2] = (unsigned)ga;                              // global_addr[31:0]
  g0[3] = (unsigned)((ga >> 32) & 0x01ffffffull)     // global_addr[56:32]
          | (2u << 30);                              // type = 2 ("image")
  v8i g1;
  g1[0] = (int)(1u << 16);                           // data_size=1 -> 2 bytes
  g1[1] = (int)((tensor_d0 & 0xffffu) << 16);        // tensor_dim0[15:0]
  g1[2] = (int)((tensor_d0 >> 16) | ((tensor_d1 & 0xffffu) << 16));
  g1[3] = (int)((tensor_d1 >> 16) | (tile_d0 << 16));// tile_dim0
  g1[4] = (int)(tile_d1 & 0xffffu);                  // tile_dim1, tile_dim2=0
  g1[5] = (int)stride0;                              // tensor_dim0_stride lo
  g1[6] = 0;                                         // stride hi / dim1_stride
  g1[7] = 0;
  const v4i zz = {};
#if TDM_6ARG
  const v8i z8 = {};
  __builtin_amdgcn_tensor_load_to_lds(g0, g1, zz, zz, z8, 0);
#else
  __builtin_amdgcn_tensor_load_to_lds(g0, g1, zz, zz, 0);
#endif
}
#endif // USE_TDM

// ---------------------------------------------------------------------------
// WMMA helpers (CDNA5, wave32, 16x16x32 f16 -> f32)
// ---------------------------------------------------------------------------
DINL v8f wmma_f16(v16h a, v16h b, v8f c) {
  return __builtin_amdgcn_wmma_f32_16x16x32_f16(false, a, false, b, (short)0, c,
                                                false, false);
}

// A fragment 16(M)x32(K): lane m = lane&15, kb = lane<16?0:16,
// elems 0..15 -> K=kb..kb+15 (contiguous, matches B fragments)
DINL v16h a_frag_f32(const float* __restrict__ A, int lda, int row0, int k0, int lane) {
  const int m  = lane & 15;
  const int kb = (lane < 16) ? 0 : 16;
  const float* p = A + (size_t)(row0 + m) * lda + k0 + kb;
  v16h r;
#pragma unroll
  for (int j = 0; j < 8; ++j) r[j] = (_Float16)p[j];
#pragma unroll
  for (int j = 0; j < 8; ++j) r[8 + j] = (_Float16)p[8 + j];
  return r;
}

DINL v16h a_frag_f16(const _Float16* __restrict__ A, int lda, int row0, int k0, int lane) {
  const int m  = lane & 15;
  const int kb = (lane < 16) ? 0 : 16;
  const _Float16* p = A + (size_t)(row0 + m) * lda + k0 + kb;
  v8h lo = *(const v8h*)p;          // 16B aligned
  v8h hi = *(const v8h*)(p + 8);    // 16B aligned
  return __builtin_shufflevector(lo, hi, 0, 1, 2, 3, 4, 5, 6, 7,
                                 8, 9, 10, 11, 12, 13, 14, 15);
}

// B fragment 32(K)x16(N), B[k][n] = W[n][k] (row-major W[N,K], i.e. A@W^T)
DINL v16h b_frag_wt_f32(const float* __restrict__ W, int ldw, int n0, int k0, int lane) {
  const int n  = lane & 15;
  const int kb = (lane < 16) ? 0 : 16;
  const float* p = W + (size_t)(n0 + n) * ldw + k0 + kb;
  v16h r;
#pragma unroll
  for (int j = 0; j < 16; ++j) r[j] = (_Float16)p[j];
  return r;
}

// Same but f16 source: contiguous 32B -> vector load
DINL v16h b_frag_wt_f16(const _Float16* __restrict__ W, int ldw, int n0, int k0, int lane) {
  const int n  = lane & 15;
  const int kb = (lane < 16) ? 0 : 16;
  const _Float16* p = W + (size_t)(n0 + n) * ldw + k0 + kb;
  return *(const v16h*)p;           // 32B aligned
}

// B fragment from an LDS-staged [32][64] f16 panel: B[k][n] = buf[k][n]
DINL v16h b_frag_lds(const _Float16* buf, int n0, int lane) {
  const int n  = n0 + (lane & 15);
  const int kb = (lane < 16) ? 0 : 16;
  v16h r;
#pragma unroll
  for (int j = 0; j < 16; ++j) r[j] = buf[(kb + j) * kHD + n];
  return r;
}

// ---------------------------------------------------------------------------
// Kernel 1: Y = X @ W^T + b, f16 output in per-head layout [B,H,S,HD].
// One wave per 16(M) x 64(N) tile: shared A-fragment, 4 WMMAs per K-step.
// ---------------------------------------------------------------------------
__global__ void proj_to_heads_kernel(const float* __restrict__ X,
                                     const float* __restrict__ W,
                                     const float* __restrict__ bias,
                                     _Float16* __restrict__ Out) {
  const int lane = threadIdx.x & 31;
  const int wave = (int)((blockIdx.x * blockDim.x + threadIdx.x) >> 5);
  const int ng = wave & 15;          // 16 groups of 64 columns
  const int mt = wave >> 4;          // 512 M tiles
  if (mt >= kM / 16) return;
  const int row0 = mt * 16, n0 = ng * 64;

  v8f acc[4] = {{}, {}, {}, {}};
  for (int k0 = 0; k0 < kHID; k0 += 32) {
    if (k0 + 64 <= kHID)
      __builtin_prefetch(X + (size_t)(row0 + (lane & 15)) * kHID + k0 + 64, 0, 0);
    const v16h a = a_frag_f32(X, kHID, row0, k0, lane);
#pragma unroll
    for (int c = 0; c < 4; ++c)
      acc[c] = wmma_f16(a, b_frag_wt_f32(W, kHID, n0 + c * 16, k0, lane), acc[c]);
  }

  const int moff = (lane < 16) ? 0 : 8;
#pragma unroll
  for (int c = 0; c < 4; ++c) {
    const int n  = n0 + c * 16 + (lane & 15);
    const float bn = bias[n];
    const int h = n >> 6, d = n & 63;
#pragma unroll
    for (int i = 0; i < 8; ++i) {
      const int m = row0 + moff + i;
      Out[(((size_t)(m >> 11) * kH + h) * kS + (m & (kS - 1))) * kHD + d] =
          (_Float16)(acc[c][i] + bn);
    }
  }
}

// ---------------------------------------------------------------------------
// Kernel 2: fused energy + mask + softmax.
// One 256-thread block owns a 16(q) x 2048(k) strip of one (b,h); energy is
// WMMA'd into 128 KB LDS, softmaxed in LDS, final attention written once.
// ---------------------------------------------------------------------------
__global__ void energy_softmax_kernel(const _Float16* __restrict__ Qh,
                                      const _Float16* __restrict__ Kh,
                                      const int* __restrict__ mask,
                                      float* __restrict__ Attn) {
  __shared__ float E[16][kS];        // 128 KB of the WGP's 320 KB
  const int t    = threadIdx.x;
  const int lane = t & 31;
  const int wv   = t >> 5;           // 8 waves
  const int blk  = blockIdx.x;       // 64*128 blocks
  const int bh   = blk >> 7;
  const int qt   = blk & 127;

  const _Float16* Q = Qh + (size_t)bh * kS * kHD;
  const _Float16* K = Kh + (size_t)bh * kS * kHD;
  const int* mrow   = mask + (bh >> 4) * kS;

  const v16h a0 = a_frag_f16(Q, kHD, qt * 16, 0, lane);
  const v16h a1 = a_frag_f16(Q, kHD, qt * 16, 32, lane);

  const int nloc = lane & 15;
  const int moff = (lane < 16) ? 0 : 8;
#pragma unroll 4
  for (int it = 0; it < 16; ++it) {
    const int kt = wv + it * 8;
    v8f acc = {};
    v16h b0 = b_frag_wt_f16(K, kHD, kt * 16, 0, lane);
    v16h b1 = b_frag_wt_f16(K, kHD, kt * 16, 32, lane);
    acc = wmma_f16(a0, b0, acc);
    acc = wmma_f16(a1, b1, acc);
    const int n = kt * 16 + nloc;
    const bool mz = (mrow[n] == 0);
#pragma unroll
    for (int i = 0; i < 8; ++i) {
      const float e = acc[i] * 0.125f;     // 1/sqrt(HD)
      E[moff + i][n] = mz ? -1.0e10f : e;
    }
  }
  __syncthreads();

  // Row softmax: 16 threads per row (half-wave -> shfl width 16)
  const int row = t >> 4;
  const int cid = t & 15;
  float mx = -3.0e38f;
  for (int i = 0; i < kS / 16; ++i) mx = fmaxf(mx, E[row][cid + i * 16]);
#pragma unroll
  for (int off = 8; off >= 1; off >>= 1) mx = fmaxf(mx, __shfl_xor(mx, off, 16));

  float s = 0.0f;
  for (int i = 0; i < kS / 16; ++i) {
    const float v = __expf(E[row][cid + i * 16] - mx);
    E[row][cid + i * 16] = v;
    s += v;
  }
#pragma unroll
  for (int off = 8; off >= 1; off >>= 1) s += __shfl_xor(s, off, 16);
  const float inv = 1.0f / s;

  float* out = Attn + (size_t)bh * kS * kS + (size_t)(qt * 16 + row) * kS;
  for (int i = 0; i < kS / 16; ++i) out[cid + i * 16] = E[row][cid + i * 16] * inv;
}

// ---------------------------------------------------------------------------
// Kernel 3: x = attn @ V, written f16 in the reference's scrambled layout
// X2[b, d*32 + h*2 + q1, q0]  (q = q1*1024 + q0).
// Block = one (b,h) x 128 q-rows; the 32x64 V panel chunk is staged to LDS by
// the Tensor Data Mover (double-buffered, DMA overlapped with WMMA).
// Each wave: 16(q) x 64(d) tile, 4 WMMAs per K-step over Sk=2048.
// ---------------------------------------------------------------------------
__global__ void av_kernel(const float* __restrict__ Attn,
                          const _Float16* __restrict__ Vh,
                          _Float16* __restrict__ X2) {
  __shared__ _Float16 vbuf[2][32 * kHD];   // 2 x 4 KB
  const int t    = threadIdx.x;
  const int lane = t & 31;
  const int wv   = t >> 5;                 // 8 waves
  const int bh   = blockIdx.x >> 4;        // 64 (b,h) slices
  const int qg   = blockIdx.x & 15;        // 16 q groups
  const int qt   = qg * 8 + wv;            // this wave's q tile (0..127)

  const float* A    = Attn + (size_t)bh * kS * kS + (size_t)(qt * 16) * kS;
  const _Float16* V = Vh + (size_t)bh * kS * kHD;

  v8f acc[4] = {{}, {}, {}, {}};

  // Stage chunk 0
#if USE_TDM
  if (wv == 0)
    tdm_load_tile_f16((unsigned)(uintptr_t)&vbuf[0][0], V,
                      kHD, kS, kHD, 32, kHD);
#else
  *(v8h*)(&vbuf[0][0] + t * 8) = *(const v8h*)(V + t * 8);
#endif

  for (int k0 = 0; k0 < kS; k0 += 32) {
    const int cur = (k0 >> 5) & 1;
#if USE_TDM
    if (wv == 0) __builtin_amdgcn_s_wait_tensorcnt(0);
#endif
    __syncthreads();                       // vbuf[cur] ready for everyone

    if (k0 + 32 < kS) {                    // overlap next chunk's DMA
#if USE_TDM
      if (wv == 0)
        tdm_load_tile_f16((unsigned)(uintptr_t)&vbuf[cur ^ 1][0],
                          V + (size_t)(k0 + 32) * kHD,
                          kHD, kS, kHD, 32, kHD);
#else
      *(v8h*)(&vbuf[cur ^ 1][0] + t * 8) =
          *(const v8h*)(V + (size_t)(k0 + 32) * kHD + t * 8);
#endif
    }

    if (k0 + 64 <= kS)
      __builtin_prefetch(A + (size_t)(lane & 15) * kS + k0 + 64, 0, 0);
    const v16h a = a_frag_f32(A, kS, 0, k0, lane);   // fp32 attn -> f16
#pragma unroll
    for (int c = 0; c < 4; ++c)
      acc[c] = wmma_f16(a, b_frag_lds(&vbuf[cur][0], c * 16, lane), acc[c]);

    __syncthreads();                       // all reads of vbuf[cur] done
  }

  const int b_ = bh >> 4, h = bh & 15;
  const int moff = (lane < 16) ? 0 : 8;
#pragma unroll
  for (int c = 0; c < 4; ++c) {
    const int d = c * 16 + (lane & 15);
#pragma unroll
    for (int i = 0; i < 8; ++i) {
      const int q = qt * 16 + moff + i;
      X2[((size_t)b_ * kS + (d * 32 + h * 2 + (q >> 10))) * kHID + (q & 1023)] =
          (_Float16)acc[c][i];
    }
  }
}

// ---------------------------------------------------------------------------
// Kernel 4: out = X2 @ Wo^T + bo (fp32, first part of d_out).
// One wave per 16(M) x 64(N) tile, 4 WMMAs per K-step.
// ---------------------------------------------------------------------------
__global__ void out_proj_kernel(const _Float16* __restrict__ X2,
                                const float* __restrict__ Wo,
                                const float* __restrict__ bo,
                                float* __restrict__ Out) {
  const int lane = threadIdx.x & 31;
  const int wave = (int)((blockIdx.x * blockDim.x + threadIdx.x) >> 5);
  const int ng = wave & 15;
  const int mt = wave >> 4;
  if (mt >= kM / 16) return;
  const int row0 = mt * 16, n0 = ng * 64;

  v8f acc[4] = {{}, {}, {}, {}};
  for (int k0 = 0; k0 < kHID; k0 += 32) {
    const v16h a = a_frag_f16(X2, kHID, row0, k0, lane);
#pragma unroll
    for (int c = 0; c < 4; ++c)
      acc[c] = wmma_f16(a, b_frag_wt_f32(Wo, kHID, n0 + c * 16, k0, lane), acc[c]);
  }

  const int moff = (lane < 16) ? 0 : 8;
#pragma unroll
  for (int c = 0; c < 4; ++c) {
    const int n = n0 + c * 16 + (lane & 15);
    const float bn = bo[n];
#pragma unroll
    for (int i = 0; i < 8; ++i) {
      const int m = row0 + moff + i;
      Out[(size_t)m * kHID + n] = acc[c][i] + bn;
    }
  }
}

// ---------------------------------------------------------------------------
// Host launcher
// ---------------------------------------------------------------------------
extern "C" void kernel_launch(void* const* d_in, const int* in_sizes, int n_in,
                              void* d_out, int out_size, void* d_ws, size_t ws_size,
                              hipStream_t stream) {
  (void)in_sizes; (void)n_in; (void)out_size; (void)ws_size;

  const float* query = (const float*)d_in[0];
  const float* key   = (const float*)d_in[1];
  const float* value = (const float*)d_in[2];
  const int*   mask  = (const int*)d_in[3];
  const float* Wq = (const float*)d_in[4];
  const float* bq = (const float*)d_in[5];
  const float* Wk = (const float*)d_in[6];
  const float* bk = (const float*)d_in[7];
  const float* Wv = (const float*)d_in[8];
  const float* bv = (const float*)d_in[9];
  const float* Wo = (const float*)d_in[10];
  const float* bo = (const float*)d_in[11];

  float* out  = (float*)d_out;            // [B, 2048, 1024] fp32
  float* attn = out + kQKV;               // [B, H, 2048, 2048] fp32 (output #2)

  _Float16* Qh = (_Float16*)d_ws;         // [B,H,S,HD] f16
  _Float16* Kh = Qh + kQKV;
  _Float16* Vh = Kh + kQKV;
  _Float16* X2 = Vh + kQKV;               // [B, 2048, 1024] scrambled merge

  const dim3 blk(256);
  // 8192 waves / 8 per block
  proj_to_heads_kernel<<<1024, blk, 0, stream>>>(query, Wq, bq, Qh);
  proj_to_heads_kernel<<<1024, blk, 0, stream>>>(key,   Wk, bk, Kh);
  proj_to_heads_kernel<<<1024, blk, 0, stream>>>(value, Wv, bv, Vh);
  // one block per (b,h,qtile): 64 * 128
  energy_softmax_kernel<<<8192, blk, 0, stream>>>(Qh, Kh, mask, attn);
  // one block per (b,h,qgroup): 64 * 16
  av_kernel<<<1024, blk, 0, stream>>>(attn, Vh, X2);
  out_proj_kernel<<<1024, blk, 0, stream>>>(X2, Wo, bo, out);
}
